// MyDataLayer_79216376807649
// MI455X (gfx1250) — compile-verified, matrix operand order checked by reading, not run on hardware
//
#include <hip/hip_runtime.h>

// ---------------------------------------------------------------------------
// CDNA5 (gfx1250) wave32 WMMA types
// ---------------------------------------------------------------------------
typedef __attribute__((ext_vector_type(16))) __bf16 v16bf;
typedef __attribute__((ext_vector_type(8)))  __bf16 v8bf;
typedef __attribute__((ext_vector_type(8)))  float  v8f;

// Problem constants
#define BB 32
#define TT 2048
#define FF 256
#define HH 128
#define G3 384   // 3*H

__device__ __forceinline__ float sigmoidf_fast(float x) {
  x = fminf(fmaxf(x, -30.f), 30.f);
  return 1.f / (1.f + __expf(-x));
}
__device__ __forceinline__ float tanhf_fast(float x) {
  x = fminf(fmaxf(x, -15.f), 15.f);
  float e = __expf(-2.f * x);
  return (1.f - e) / (1.f + e);
}

// ---------------------------------------------------------------------------
// Kernel 0: pack f32 weights -> bf16 in WMMA B-operand fragment layout.
// B operand of v_wmma_f32_16x16x32_bf16 is a 32(K)x16(N) tile; lane l holds
// column n = l&15, with k-base kb = (l>>4)*8:
//   frag elements e=0..7  -> K = K0 + kb + e
//   frag elements e=8..15 -> K = K0 + 16 + kb + (e-8)
// We store each lane's 16 bf16 contiguously so the GEMM does one 32B load.
// Tile order: [nt][kc][lane][16]   (kc contiguous for the K-loop)
// ---------------------------------------------------------------------------
__global__ void pack_weights(const float* __restrict__ Wd,   // 256x128
                             const float* __restrict__ Wg,   // 256x384
                             const float* __restrict__ Wr,   // 128x384
                             __bf16* __restrict__ pWd,
                             __bf16* __restrict__ pWg,
                             __bf16* __restrict__ pWr) {
  int tid = blockIdx.x * blockDim.x + threadIdx.x;   // 11264 total
  const float* src; __bf16* dst; int N, idx, ktiles;
  if (tid < 2048)      { src = Wd; dst = pWd; N = 128; idx = tid;        ktiles = 8; }
  else if (tid < 8192) { src = Wg; dst = pWg; N = 384; idx = tid - 2048; ktiles = 8; }
  else                 { src = Wr; dst = pWr; N = 384; idx = tid - 8192; ktiles = 4; }
  int lane = idx & 31;
  int rem  = idx >> 5;
  int kc   = rem % ktiles;
  int nt   = rem / ktiles;
  int n    = lane & 15;
  int kb   = (lane >> 4) * 8;
  int K0   = kc * 32;
  int nabs = nt * 16 + n;
  __bf16* d = dst + (size_t)((nt * ktiles + kc) * 32 + lane) * 16;
#pragma unroll
  for (int e = 0; e < 16; ++e) {
    int k = K0 + ((e < 8) ? (kb + e) : (16 + kb + (e - 8)));
    d[e] = (__bf16)src[(size_t)k * N + nabs];
  }
}

// ---------------------------------------------------------------------------
// Kernel 1: fused front GEMMs.
// One wave per 16-row stripe of the [B*T, 256] activations.
//   lit  = relu(literal @ W_dense)        -> out[:, :, 0:128]
//   xp   = semantic @ gru_kernel + b_in   -> ws xp buffer, [T][B][384] layout
// A operand (16x32 bf16): lane l holds row m = l&15, kb = (l>>4)*8,
//   elements 0..7 = K0+kb..+7, elements 8..15 = K0+16+kb..+7 (contiguous f32
//   runs in row-major memory -> cheap load+cvt).
// ---------------------------------------------------------------------------
__global__ __launch_bounds__(256) void gemm_front(
    const float* __restrict__ lit, const float* __restrict__ sem,
    const __bf16* __restrict__ pWd, const __bf16* __restrict__ pWg,
    const float* __restrict__ b_in,
    float* __restrict__ out, float* __restrict__ xp) {
  int lane   = threadIdx.x & 31;
  int wgid   = (blockIdx.x * blockDim.x + threadIdx.x) >> 5;   // 0..4095
  int r0     = wgid * 16;
  int m      = lane & 15;
  int kb     = (lane >> 4) * 8;
  int n      = lane & 15;
  int rowoff = (lane >> 4) * 8;   // C layout: VGPR r holds rows r (lanes<16) / r+8

  // ---- literal branch -----------------------------------------------------
  {
    const float* abase = lit + (size_t)(r0 + m) * FF;
    v16bf A[8];
#pragma unroll
    for (int kc = 0; kc < 8; ++kc) {
      const float* p = abase + kc * 32 + kb;
      v16bf a;
#pragma unroll
      for (int e = 0; e < 8; ++e) { a[e] = (__bf16)p[e]; a[8 + e] = (__bf16)p[16 + e]; }
      A[kc] = a;
    }
    for (int nt = 0; nt < 8; ++nt) {
      v8f acc = {0.f, 0.f, 0.f, 0.f, 0.f, 0.f, 0.f, 0.f};
#pragma unroll
      for (int kc = 0; kc < 8; ++kc) {
        v16bf bm = *(const v16bf*)(pWd + (size_t)((nt * 8 + kc) * 32 + lane) * 16);
        acc = __builtin_amdgcn_wmma_f32_16x16x32_bf16(
            false, A[kc], false, bm, (short)0, acc, false, false);
      }
      int col = nt * 16 + n;
#pragma unroll
      for (int r = 0; r < 8; ++r) {
        size_t Row = (size_t)(r0 + rowoff + r);
        out[Row * 256 + col] = fmaxf(acc[r], 0.f);
      }
    }
  }
  // ---- semantic -> xp branch ---------------------------------------------
  {
    const float* abase = sem + (size_t)(r0 + m) * FF;
    v16bf A[8];
#pragma unroll
    for (int kc = 0; kc < 8; ++kc) {
      const float* p = abase + kc * 32 + kb;
      v16bf a;
#pragma unroll
      for (int e = 0; e < 8; ++e) { a[e] = (__bf16)p[e]; a[8 + e] = (__bf16)p[16 + e]; }
      A[kc] = a;
    }
    for (int nt = 0; nt < 24; ++nt) {
      v8f acc = {0.f, 0.f, 0.f, 0.f, 0.f, 0.f, 0.f, 0.f};
#pragma unroll
      for (int kc = 0; kc < 8; ++kc) {
        v16bf bm = *(const v16bf*)(pWg + (size_t)((nt * 8 + kc) * 32 + lane) * 16);
        acc = __builtin_amdgcn_wmma_f32_16x16x32_bf16(
            false, A[kc], false, bm, (short)0, acc, false, false);
      }
      int col = nt * 16 + n;
      float bias = b_in[col];
#pragma unroll
      for (int r = 0; r < 8; ++r) {
        int Row  = r0 + rowoff + r;
        int t    = Row & (TT - 1);
        int bidx = Row >> 11;
        xp[((size_t)t * BB + bidx) * G3 + col] = acc[r] + bias;
      }
    }
  }
}

// ---------------------------------------------------------------------------
// Kernel 2: sequential GRU scan. Batch is independent -> 2 blocks x 16 batch
// rows. h kept in LDS (f32 state + bf16 WMMA mirror). Per step:
//   inner = h @ rec_kernel  (1 M-tile x 24 N-tiles x 4 K-chunks = 96 WMMA,
//   3 tiles per wave across 8 waves) -> LDS, barrier, vector gate pass,
//   barrier. Next step's xp slice is prefetched (global_prefetch_b8).
// ---------------------------------------------------------------------------
__global__ __launch_bounds__(256) void gru_seq(
    const float* __restrict__ xp, const __bf16* __restrict__ pWr,
    const float* __restrict__ b_rec, float* __restrict__ out) {
  __shared__ float  s_inner[16 * G3];   // 24 KB
  __shared__ float  s_hf[16 * HH];      //  8 KB
  __shared__ __bf16 s_hb[16 * HH];      //  4 KB

  int tid    = threadIdx.x;
  int lane   = tid & 31;
  int wave   = tid >> 5;                 // 0..7
  int m      = lane & 15;
  int kb     = (lane >> 4) * 8;
  int n      = lane & 15;
  int rowoff = (lane >> 4) * 8;
  int bloc   = tid >> 4;                 // local batch row 0..15 (gate pass)
  int n0     = (tid & 15) * 8;           // 8 h-elements per thread
  int gbatch = blockIdx.x * 16 + bloc;   // global batch row

  for (int i = tid; i < 16 * HH; i += 256) { s_hf[i] = 0.f; s_hb[i] = (__bf16)0.f; }

  float bz[8], br[8], bh[8];
#pragma unroll
  for (int e = 0; e < 8; ++e) {
    bz[e] = b_rec[n0 + e];
    br[e] = b_rec[HH + n0 + e];
    bh[e] = b_rec[2 * HH + n0 + e];
  }
  __syncthreads();

  for (int t = 0; t < TT; ++t) {
    // ---- inner = h @ rec_kernel ----
#pragma unroll
    for (int j = 0; j < 3; ++j) {
      int nt = wave * 3 + j;             // 0..23
      v8f acc = {0.f, 0.f, 0.f, 0.f, 0.f, 0.f, 0.f, 0.f};
#pragma unroll
      for (int kc = 0; kc < 4; ++kc) {
        const __bf16* hrow = &s_hb[m * HH + kc * 32];
        v8bf lo = *(const v8bf*)(hrow + kb);
        v8bf hi = *(const v8bf*)(hrow + 16 + kb);
        v16bf a = __builtin_shufflevector(lo, hi, 0, 1, 2, 3, 4, 5, 6, 7,
                                          8, 9, 10, 11, 12, 13, 14, 15);
        v16bf bm = *(const v16bf*)(pWr + (size_t)((nt * 4 + kc) * 32 + lane) * 16);
        acc = __builtin_amdgcn_wmma_f32_16x16x32_bf16(
            false, a, false, bm, (short)0, acc, false, false);
      }
      int col = nt * 16 + n;
#pragma unroll
      for (int r = 0; r < 8; ++r)
        s_inner[(rowoff + r) * G3 + col] = acc[r];
    }
    __syncthreads();

    // ---- gate pass ----
    const float* xpt = xp + ((size_t)t * BB + gbatch) * G3;
#pragma unroll
    for (int e = 0; e < 8; ++e) {
      int nn = n0 + e;
      float xz = xpt[nn], xr = xpt[HH + nn], xh = xpt[2 * HH + nn];
      float iz = s_inner[bloc * G3 + nn] + bz[e];
      float ir = s_inner[bloc * G3 + HH + nn] + br[e];
      float ih = s_inner[bloc * G3 + 2 * HH + nn] + bh[e];
      float z  = sigmoidf_fast(xz + iz);
      float rg = sigmoidf_fast(xr + ir);
      float hh = tanhf_fast(xh + rg * ih);
      float ho = s_hf[bloc * HH + nn];
      float hn = z * ho + (1.f - z) * hh;
      s_hf[bloc * HH + nn] = hn;
      s_hb[bloc * HH + nn] = (__bf16)hn;
      out[((size_t)gbatch * TT + t) * 256 + HH + nn] = hn;
    }
    if (t + 1 < TT) {
      const float* nx = xp + ((size_t)(t + 1) * BB + gbatch) * G3;
      __builtin_prefetch(nx + n0, 0, 0);
      __builtin_prefetch(nx + HH + n0, 0, 0);
      __builtin_prefetch(nx + 2 * HH + n0, 0, 0);
    }
    __syncthreads();
  }
}

// ---------------------------------------------------------------------------
// Launch: pack -> front GEMMs -> GRU scan (stream-ordered dependencies).
// Workspace layout: [0,64K) pWd | [64K,256K) pWg | [256K,352K) pWr |
//                   [512K, 512K+96MB) xp  (assumes ws_size is ample)
// ---------------------------------------------------------------------------
extern "C" void kernel_launch(void* const* d_in, const int* in_sizes, int n_in,
                              void* d_out, int out_size, void* d_ws, size_t ws_size,
                              hipStream_t stream) {
  (void)in_sizes; (void)n_in; (void)out_size; (void)ws_size;
  const float* literal  = (const float*)d_in[0];
  const float* semantic = (const float*)d_in[1];
  const float* W_dense  = (const float*)d_in[2];
  const float* gru_k    = (const float*)d_in[3];
  const float* gru_rk   = (const float*)d_in[4];
  const float* gru_bin  = (const float*)d_in[5];
  const float* gru_brec = (const float*)d_in[6];
  float* out = (float*)d_out;

  char* ws = (char*)d_ws;
  __bf16* pWd = (__bf16*)(ws);
  __bf16* pWg = (__bf16*)(ws + 65536);
  __bf16* pWr = (__bf16*)(ws + 262144);
  float*  xp  = (float*)(ws + 524288);

  pack_weights<<<44, 256, 0, stream>>>(W_dense, gru_k, gru_rk, pWd, pWg, pWr);
  gemm_front<<<512, 256, 0, stream>>>(literal, semantic, pWd, pWg, gru_bin, out, xp);
  gru_seq<<<2, 256, 0, stream>>>(xp, pWr, gru_brec, out);
}
